// GCNNet_pyg_59279138619792
// MI455X (gfx1250) — matrix operand order repaired and checked by reading, not compiled, not run on hardware
//
#include <hip/hip_runtime.h>
#include <hip/hip_bf16.h>

// ---------------------------------------------------------------------------
// GCN (4-layer, HID=128) for MI455X / gfx1250, wave32.
// GEMMs: v_wmma_f32_16x16x32_bf16 on PREPACKED hi/lo bf16 operands
// (split-fp32 trick: hi*hi + hi*lo + lo*hi in f32 accum ~ fp32 accuracy).
// A tiles staged in LDS via global_load_async_to_lds_b128; B prepacked
// transposed so fragments are contiguous b128 loads. Scatter is L2-resident.
// ---------------------------------------------------------------------------

#define GN 50000
#define GE 800000
#define HID 128
#define NCLS 10
#define NLAYERS 4
#define BN_EPS 1e-5f

#ifndef __has_builtin
#define __has_builtin(x) 0
#endif

#if __has_builtin(__builtin_amdgcn_global_load_async_to_lds_b128)
#define HAVE_ASYNC_LDS 1
#else
#define HAVE_ASYNC_LDS 0
#endif

typedef __attribute__((ext_vector_type(16))) __bf16 v16bf;
typedef __attribute__((ext_vector_type(8)))  __bf16 v8bf;
typedef __attribute__((ext_vector_type(8)))  float  v8f;
typedef __attribute__((ext_vector_type(4)))  int    v4i;

__device__ __forceinline__ unsigned short f32_to_bf16_rne(float f) {
  unsigned u = __builtin_bit_cast(unsigned, f);
  unsigned r = u + 0x7FFFu + ((u >> 16) & 1u);
  return (unsigned short)(r >> 16);
}
__device__ __forceinline__ float bf16bits_to_f32(unsigned short b) {
  return __builtin_bit_cast(float, ((unsigned)b) << 16);
}
__device__ __forceinline__ __bf16 bits_to_bf16(unsigned short b) {
  return __builtin_bit_cast(__bf16, b);
}
__device__ __forceinline__ void split_bf16(float a, __bf16& hi, __bf16& lo) {
  unsigned short h = f32_to_bf16_rne(a);
  float rem = a - bf16bits_to_f32(h);
  hi = bits_to_bf16(h);
  lo = bits_to_bf16(f32_to_bf16_rne(rem));
}

__device__ __forceinline__ void wait_async0() {
#if __has_builtin(__builtin_amdgcn_s_wait_asynccnt)
  __builtin_amdgcn_s_wait_asynccnt(0);
#else
  asm volatile("s_wait_asynccnt 0" ::: "memory");
#endif
}

// ---------------------------------------------------------------------------
// Prepack: split fp32 matrix A[N x 128] into bf16 hi/lo arrays (row-major).
// ---------------------------------------------------------------------------
__global__ void pack_A_kernel(const float* __restrict__ A,
                              __bf16* __restrict__ hi,
                              __bf16* __restrict__ lo) {
  const size_t base = ((size_t)blockIdx.x * blockDim.x + threadIdx.x) * 4;
  const float4 v = *(const float4*)(A + base);
  const float vv[4] = {v.x, v.y, v.z, v.w};
#pragma unroll
  for (int j = 0; j < 4; ++j) {
    __bf16 h, l;
    split_bf16(vv[j], h, l);
    hi[base + j] = h;
    lo[base + j] = l;
  }
}

// ---------------------------------------------------------------------------
// Prepack weights: W[128 x ncols] (row-major) -> transposed, zero-padded,
// split:  WThi/WTlo[col][k]  (col in [0, ntpad), k in [0,128)).
// Makes each lane's B fragment two contiguous runs of 8 bf16 (two b128 loads).
// ---------------------------------------------------------------------------
__global__ void pack_WT_kernel(const float* __restrict__ W,
                               __bf16* __restrict__ WThi,
                               __bf16* __restrict__ WTlo,
                               int ncols) {
  const int col = blockIdx.x;     // 0..ntpad-1
  const int k   = threadIdx.x;    // 0..127
  const float v = (col < ncols) ? W[(size_t)k * ncols + col] : 0.0f;
  __bf16 h, l;
  split_bf16(v, h, l);
  WThi[(size_t)col * HID + k] = h;
  WTlo[(size_t)col * HID + k] = l;
}

// ---------------------------------------------------------------------------
// C[M x ncols] = A @ W + bias from prepacked operands. One wave per 16x16
// tile, 8 waves/block = one 16-row x 128-col stripe. A tile (8KB) staged in
// LDS via async-to-LDS; 4 K-steps x 3 WMMAs (hi*hi, hi*lo, lo*hi).
// ---------------------------------------------------------------------------
__global__ void gemm_packed_kernel(const __bf16* __restrict__ Ahi,
                                   const __bf16* __restrict__ Alo,
                                   const __bf16* __restrict__ WThi,
                                   const __bf16* __restrict__ WTlo,
                                   const float* __restrict__ bias,
                                   float* __restrict__ C,
                                   int ncols) {
  __shared__ __align__(16) __bf16 sAhi[16 * HID];
  __shared__ __align__(16) __bf16 sAlo[16 * HID];

  const int tid = threadIdx.x;          // 0..255
  const int m0  = blockIdx.x * 16;

  // ---- stage the 16x128 hi/lo A tile (8KB) into LDS: 16B per thread/array
  {
    const size_t g = (size_t)m0 * HID + (size_t)tid * 8;
#if HAVE_ASYNC_LDS
    __builtin_amdgcn_global_load_async_to_lds_b128(
        (v4i*)(Ahi + g), (v4i*)(sAhi + tid * 8), 0, 0);
    __builtin_amdgcn_global_load_async_to_lds_b128(
        (v4i*)(Alo + g), (v4i*)(sAlo + tid * 8), 0, 0);
    wait_async0();
#else
    *(v8bf*)(sAhi + tid * 8) = *(const v8bf*)(Ahi + g);
    *(v8bf*)(sAlo + tid * 8) = *(const v8bf*)(Alo + g);
#endif
  }
  __syncthreads();                      // all 8 waves reach this, always

  const int wave  = tid >> 5;           // 0..7 -> column tile
  const int n0    = wave * 16;
  const int ntpad = (ncols + 15) & ~15;
  if (n0 >= ntpad) return;              // wave-uniform, after the barrier

  const int lane = tid & 31;
  const int half = lane >> 4;           // 0: lanes 0-15, 1: lanes 16-31
  const int l15  = lane & 15;
  const int col  = n0 + l15;            // B/C column held by this lane

  const __bf16* __restrict__ bhiCol = WThi + (size_t)col * HID;
  const __bf16* __restrict__ bloCol = WTlo + (size_t)col * HID;

  v8f acc = {};
#pragma unroll
  for (int kk = 0; kk < 4; ++kk) {
    const int kb = kk * 32;

    // A fragment (ISA 7.12.2): elems 0-7 <- K=kb+half*8+(0..7),
    //                          elems 8-15 <- K=kb+16+half*8+(0..7)
    const int aoff = l15 * HID + kb + half * 8;
    const v8bf ah0 = *(const v8bf*)(sAhi + aoff);
    const v8bf ah1 = *(const v8bf*)(sAhi + aoff + 16);
    const v8bf al0 = *(const v8bf*)(sAlo + aoff);
    const v8bf al1 = *(const v8bf*)(sAlo + aoff + 16);
    const v16bf ahi = __builtin_shufflevector(ah0, ah1, 0, 1, 2, 3, 4, 5, 6, 7,
                                              8, 9, 10, 11, 12, 13, 14, 15);
    const v16bf alo = __builtin_shufflevector(al0, al1, 0, 1, 2, 3, 4, 5, 6, 7,
                                              8, 9, 10, 11, 12, 13, 14, 15);

    // B fragment: elems 0-7 <- K = kb + half*8 + (0..7),
    //             elems 8-15 <- K = kb + 16 + half*8 + (0..7)
    const int boff = kb + half * 8;
    const v8bf bh0 = *(const v8bf*)(bhiCol + boff);
    const v8bf bh1 = *(const v8bf*)(bhiCol + boff + 16);
    const v8bf bl0 = *(const v8bf*)(bloCol + boff);
    const v8bf bl1 = *(const v8bf*)(bloCol + boff + 16);
    const v16bf bhi = __builtin_shufflevector(bh0, bh1, 0, 1, 2, 3, 4, 5, 6, 7,
                                              8, 9, 10, 11, 12, 13, 14, 15);
    const v16bf blo = __builtin_shufflevector(bl0, bl1, 0, 1, 2, 3, 4, 5, 6, 7,
                                              8, 9, 10, 11, 12, 13, 14, 15);

    acc = __builtin_amdgcn_wmma_f32_16x16x32_bf16(false, ahi, false, bhi,
                                                  (short)0, acc, false, false);
    acc = __builtin_amdgcn_wmma_f32_16x16x32_bf16(false, ahi, false, blo,
                                                  (short)0, acc, false, false);
    acc = __builtin_amdgcn_wmma_f32_16x16x32_bf16(false, alo, false, bhi,
                                                  (short)0, acc, false, false);
  }

  if (col < ncols) {
    const float bv = bias ? bias[col] : 0.0f;
#pragma unroll
    for (int r = 0; r < 8; ++r) {
      const int m = m0 + r + 8 * half;  // C/D layout: VGPR r, half
      C[(size_t)m * ncols + col] = acc[r] + bv;
    }
  }
}

// ---------------------------------------------------------------------------
// gcn_norm pieces
// ---------------------------------------------------------------------------
__global__ void init_deg_kernel(float* __restrict__ deg) {
  int i = blockIdx.x * blockDim.x + threadIdx.x;
  if (i < GN) deg[i] = 1.0f;              // self-loop contribution
}

__global__ void scatter_deg_kernel(const long long* __restrict__ dst,
                                   float* __restrict__ deg) {
  int e = blockIdx.x * blockDim.x + threadIdx.x;
  if (e < GE) atomicAdd(&deg[dst[e]], 1.0f);
}

__global__ void finalize_dinv_kernel(float* __restrict__ deg) {
  int i = blockIdx.x * blockDim.x + threadIdx.x;
  if (i < GN) deg[i] = rsqrtf(deg[i]);    // deg >= 1 always (self-loop)
}

// agg[n][c] = bias[c] + dinv[n]^2 * x[n][c]   (self-loop message + bias)
__global__ void agg_init_kernel(const float* __restrict__ x,
                                const float* __restrict__ dinv,
                                const float* __restrict__ bias,
                                float* __restrict__ agg) {
  const int n = blockIdx.x;
  const int c = threadIdx.x;
  const float di = dinv[n];
  agg[(size_t)n * HID + c] = bias[c] + di * di * x[(size_t)n * HID + c];
}

// ---------------------------------------------------------------------------
// One wave32 per edge: agg[dst] += dinv[src]*dinv[dst] * x[src]
// float4 (b128) coalesced reads, 4 f32 atomics per lane. L2-resident.
// ---------------------------------------------------------------------------
__global__ void scatter_msg_kernel(const long long* __restrict__ src,
                                   const long long* __restrict__ dst,
                                   const float* __restrict__ dinv,
                                   const float* __restrict__ x,
                                   float* __restrict__ agg) {
  const int gid  = blockIdx.x * blockDim.x + threadIdx.x;
  const int e    = gid >> 5;
  const int lane = gid & 31;
  if (e >= GE) return;
  const long long s = src[e];
  const long long d = dst[e];
  const float w = dinv[s] * dinv[d];
  const float4 xv = *(const float4*)(x + (size_t)s * HID + lane * 4);
  float* __restrict__ ar = agg + (size_t)d * HID + lane * 4;
  atomicAdd(&ar[0], w * xv.x);
  atomicAdd(&ar[1], w * xv.y);
  atomicAdd(&ar[2], w * xv.z);
  atomicAdd(&ar[3], w * xv.w);
}

// ---------------------------------------------------------------------------
// BatchNorm statistics: per-block partial sums over 128 rows, then atomics.
// ---------------------------------------------------------------------------
__global__ void bn_zero_kernel(float* __restrict__ sum,
                               float* __restrict__ sumsq) {
  const int c = threadIdx.x;
  if (c < HID) { sum[c] = 0.0f; sumsq[c] = 0.0f; }
}

__global__ void bn_stats_kernel(const float* __restrict__ agg,
                                float* __restrict__ sum,
                                float* __restrict__ sumsq) {
  const int c  = threadIdx.x;            // 128 threads
  const int r0 = blockIdx.x * 128;
  const int r1 = (r0 + 128 < GN) ? (r0 + 128) : GN;
  float s = 0.0f, ss = 0.0f;
  for (int r = r0; r < r1; ++r) {
    const float v = agg[(size_t)r * HID + c];
    s += v; ss += v * v;
  }
  atomicAdd(&sum[c], s);
  atomicAdd(&sumsq[c], ss);
}

__global__ void bn_finalize_kernel(const float* __restrict__ sum,
                                   const float* __restrict__ sumsq,
                                   const float* __restrict__ gamma,
                                   const float* __restrict__ beta,
                                   float* __restrict__ scale,
                                   float* __restrict__ shift) {
  const int c = threadIdx.x;
  if (c >= HID) return;
  const float inv_n = 1.0f / (float)GN;
  const float mu  = sum[c] * inv_n;
  const float var = sumsq[c] * inv_n - mu * mu;
  const float rstd = rsqrtf(var + BN_EPS);
  const float sc = gamma[c] * rstd;
  scale[c] = sc;
  shift[c] = beta[c] - mu * sc;
}

// h[n][c] += relu(agg[n][c]*scale[c] + shift[c])   (BN + ReLU + residual)
__global__ void bn_apply_residual_kernel(const float* __restrict__ agg,
                                         const float* __restrict__ scale,
                                         const float* __restrict__ shift,
                                         float* __restrict__ h) {
  const int n = blockIdx.x;
  const int c = threadIdx.x;
  const size_t idx = (size_t)n * HID + c;
  const float v = agg[idx] * scale[c] + shift[c];
  h[idx] = h[idx] + fmaxf(v, 0.0f);
}

// ---------------------------------------------------------------------------
// Host orchestration (all launches on `stream`, graph-capture safe)
// ---------------------------------------------------------------------------
extern "C" void kernel_launch(void* const* d_in, const int* in_sizes, int n_in,
                              void* d_out, int out_size, void* d_ws, size_t ws_size,
                              hipStream_t stream) {
  const float*     h_in   = (const float*)d_in[0];          // [N,128]
  const long long* edge   = (const long long*)d_in[1];      // [2,E] int64
  /* d_in[2] = e (edge features): unused by the reference */
  const float*     W_emb  = (const float*)d_in[3];          // [128,128]
  const float*     b_emb  = (const float*)d_in[4];          // [128]
  const float*     Ws     = (const float*)d_in[5];          // [L,128,128]
  const float*     bs     = (const float*)d_in[6];          // [L,128]
  const float*     gammas = (const float*)d_in[7];          // [L,128]
  const float*     betas  = (const float*)d_in[8];          // [L,128]
  const float*     W_mlp  = (const float*)d_in[9];          // [128,10]
  const float*     b_mlp  = (const float*)d_in[10];         // [10]

  const long long* src = edge;
  const long long* dst = edge + GE;

  // Workspace carve-up. fp32 arrays first, then bf16 packs. ~103 MB total.
  float* ws    = (float*)d_ws;
  float* dinv  = ws;                         // N
  float* h     = dinv + GN;                  // N*128
  float* x     = h + (size_t)GN * HID;       // N*128
  float* agg   = x + (size_t)GN * HID;       // N*128
  float* sum   = agg + (size_t)GN * HID;     // 128
  float* sumsq = sum + HID;                  // 128
  float* scale = sumsq + HID;                // 128
  float* shift = scale + HID;                // 128
  __bf16* Ahi  = (__bf16*)(shift + HID);     // N*128
  __bf16* Alo  = Ahi + (size_t)GN * HID;     // N*128
  __bf16* WThi = Alo + (size_t)GN * HID;     // 128*128 (padded cols)
  __bf16* WTlo = WThi + HID * HID;           // 128*128
  (void)in_sizes; (void)n_in; (void)out_size; (void)ws_size;

  const int gemm_grid = GN / 16;                 // 3125, exact
  const int packA_grid = (GN * HID) / (256 * 4); // 6250, exact

  // --- gcn_norm ---
  init_deg_kernel<<<(GN + 255) / 256, 256, 0, stream>>>(dinv);
  scatter_deg_kernel<<<(GE + 255) / 256, 256, 0, stream>>>(dst, dinv);
  finalize_dinv_kernel<<<(GN + 255) / 256, 256, 0, stream>>>(dinv);

  // --- h = h_in @ W_emb + b_emb ---
  pack_A_kernel<<<packA_grid, 256, 0, stream>>>(h_in, Ahi, Alo);
  pack_WT_kernel<<<HID, HID, 0, stream>>>(W_emb, WThi, WTlo, HID);
  gemm_packed_kernel<<<gemm_grid, 256, 0, stream>>>(Ahi, Alo, WThi, WTlo,
                                                    b_emb, h, HID);

  // --- 4 GCN layers ---
  const int scat_grid = (GE * 32 + 255) / 256;   // one wave32 per edge
  for (int i = 0; i < NLAYERS; ++i) {
    const float* Wi = Ws + (size_t)i * HID * HID;
    const float* bi = bs + (size_t)i * HID;
    const float* gi = gammas + (size_t)i * HID;
    const float* ti = betas + (size_t)i * HID;

    pack_A_kernel<<<packA_grid, 256, 0, stream>>>(h, Ahi, Alo);
    pack_WT_kernel<<<HID, HID, 0, stream>>>(Wi, WThi, WTlo, HID);
    gemm_packed_kernel<<<gemm_grid, 256, 0, stream>>>(Ahi, Alo, WThi, WTlo,
                                                      nullptr, x, HID);
    agg_init_kernel<<<GN, HID, 0, stream>>>(x, dinv, bi, agg);
    scatter_msg_kernel<<<scat_grid, 256, 0, stream>>>(src, dst, dinv, x, agg);
    bn_zero_kernel<<<1, HID, 0, stream>>>(sum, sumsq);
    bn_stats_kernel<<<(GN + 127) / 128, HID, 0, stream>>>(agg, sum, sumsq);
    bn_finalize_kernel<<<1, HID, 0, stream>>>(sum, sumsq, gi, ti, scale, shift);
    bn_apply_residual_kernel<<<GN, HID, 0, stream>>>(agg, scale, shift, h);
  }

  // --- out = h @ W_mlp + b_mlp ---
  pack_A_kernel<<<packA_grid, 256, 0, stream>>>(h, Ahi, Alo);
  pack_WT_kernel<<<16, HID, 0, stream>>>(W_mlp, WThi, WTlo, NCLS);
  gemm_packed_kernel<<<gemm_grid, 256, 0, stream>>>(Ahi, Alo, WThi, WTlo,
                                                    b_mlp, (float*)d_out, NCLS);
}
